// HeteroSGCPaper_80599356276853
// MI455X (gfx1250) — compile-verified, matrix operand order checked by reading, not run on hardware
//
#include <hip/hip_runtime.h>

// CDNA5 / gfx1250: wave32, WMMA 16x16x32 bf16 (fp32 accumulate)
typedef __attribute__((ext_vector_type(16))) __bf16 v16bf;
typedef __attribute__((ext_vector_type(8)))  float  v8f;

#define ALPHA_HSGC 0.01f

__device__ __forceinline__ void atomic_add_f32(float* p, float v) {
  // relaxed agent-scope fadd -> global_atomic_add_f32
  __hip_atomic_fetch_add(p, v, __ATOMIC_RELAXED, __HIP_MEMORY_SCOPE_AGENT);
}

// ---------------- zero (float4 granularity) ----------------
__global__ __launch_bounds__(256) void zero4_kernel(float4* __restrict__ p, long n4) {
  long i = (long)blockIdx.x * blockDim.x + threadIdx.x;
  if (i < n4) p[i] = make_float4(0.f, 0.f, 0.f, 0.f);
}

// ---------------- per-dst degree count ----------------
__global__ __launch_bounds__(256) void degree_kernel(const int* __restrict__ dst,
                                                     int* __restrict__ deg, int n) {
  int i = blockIdx.x * blockDim.x + threadIdx.x;
  if (i < n) atomicAdd(&deg[dst[i]], 1);
}

// ---------------- edge scatter: agg[dst] += (w/deg[dst]) * h_src[src] ----------------
// One wave per edge; lane l moves floats [4l, 4l+3] of the 128-float row.
__global__ __launch_bounds__(256) void edge_scatter_kernel(
    const float* __restrict__ h_src, const int* __restrict__ src,
    const int* __restrict__ dst, const int* __restrict__ deg,
    float* __restrict__ agg, int n_edges, float w) {
  int wid  = (int)(((unsigned)blockIdx.x * blockDim.x + threadIdx.x) >> 5);
  int lane = threadIdx.x & 31;
  if (wid >= n_edges) return;            // wave-uniform exit
  int s = src[wid];
  int d = dst[wid];
  int dg = deg[d];
  float scale = w / (float)(dg > 0 ? dg : 1);
  if (wid + 1 < n_edges) {               // wave-uniform predicate
    // hint next edge's source row into cache (global_prefetch_b8)
    __builtin_prefetch(h_src + (size_t)src[wid + 1] * 128 + (lane << 2), 0, 3);
  }
  const float4* hs = (const float4*)(h_src + (size_t)s * 128);
  float4 v = hs[lane];
  float* out = agg + (size_t)d * 128 + (lane << 2);
  atomic_add_f32(out + 0, v.x * scale);
  atomic_add_f32(out + 1, v.y * scale);
  atomic_add_f32(out + 2, v.z * scale);
  atomic_add_f32(out + 3, v.w * scale);
}

// ---------------- combine: h = alpha*h + agg ; agg = 0 (ready for next layer) ----------------
__global__ __launch_bounds__(256) void combine_kernel(float4* __restrict__ h,
                                                      float4* __restrict__ agg,
                                                      long n4, float alpha) {
  long i = (long)blockIdx.x * blockDim.x + threadIdx.x;
  if (i >= n4) return;
  float4 a  = agg[i];
  float4 hv = h[i];
  hv.x = fmaf(alpha, hv.x, a.x);
  hv.y = fmaf(alpha, hv.y, a.y);
  hv.z = fmaf(alpha, hv.z, a.z);
  hv.w = fmaf(alpha, hv.w, a.w);
  h[i]   = hv;
  agg[i] = make_float4(0.f, 0.f, 0.f, 0.f);
}

// ---------------- WMMA GEMM: C[M,N] = act(A[M,K] @ W[N,K]^T + bias[N]) ----------------
// Requires M,N multiples of 16 and K multiple of 32 (true for this model).
// One wave computes one 16x16 output tile via v_wmma_f32_16x16x32_bf16 along K.
template <bool RELU>
__global__ __launch_bounds__(256) void gemm_bias_wmma_kernel(
    const float* __restrict__ A, const float* __restrict__ W,
    const float* __restrict__ bias, float* __restrict__ C,
    int M, int N, int K) {
  int wid = (int)(blockIdx.x * (blockDim.x >> 5) + (threadIdx.x >> 5));
  int tiles_n = N >> 4;
  int total   = (M >> 4) * tiles_n;
  if (wid >= total) return;              // wave-uniform: EXEC stays all-ones for WMMA
  int tm   = (wid / tiles_n) << 4;
  int tn   = (wid % tiles_n) << 4;
  int lane = threadIdx.x & 31;
  int r    = lane & 15;                  // A-row / B-col within tile
  int kh   = (lane >> 4) << 3;           // k-half offset: lanes 16-31 hold K+8 / K+24 pairs

  const float* Arow = A + (size_t)(tm + r) * K;   // A row (row-major, contiguous in k)
  const float* Wrow = W + (size_t)(tn + r) * K;   // B col n == W row n (W is [N,K])

  v8f acc = {};
  for (int k0 = 0; k0 < K; k0 += 32) {
    v16bf af, bf;
#pragma unroll
    for (int j = 0; j < 8; ++j) {
      // 16-bit A 16x32 layout: VGPR j<4 -> K = 2j (+kh); VGPR j>=4 -> K = 16+2(j-4) (+kh)
      int bk = k0 + ((j < 4) ? (j << 1) : (16 + ((j - 4) << 1))) + kh;
      af[2 * j]     = (__bf16)Arow[bk];
      af[2 * j + 1] = (__bf16)Arow[bk + 1];
      bf[2 * j]     = (__bf16)Wrow[bk];
      bf[2 * j + 1] = (__bf16)Wrow[bk + 1];
    }
    acc = __builtin_amdgcn_wmma_f32_16x16x32_bf16(
        /*neg_a=*/false, af, /*neg_b=*/false, bf,
        /*c_mod=*/(short)0, acc, /*reuse_a=*/false, /*reuse_b=*/false);
  }

  // D layout: VGPR j -> M = tm + j + 8*(lane>>4), N = tn + (lane&15)
  int n     = tn + r;
  int mbase = tm + ((lane >> 4) << 3);
  float bv  = bias[n];
#pragma unroll
  for (int j = 0; j < 8; ++j) {
    float v = acc[j] + bv;
    if (RELU) v = fmaxf(v, 0.f);
    C[(size_t)(mbase + j) * N + n] = v;
  }
}

static inline int cdiv_i(long a, long b) { return (int)((a + b - 1) / b); }

extern "C" void kernel_launch(void* const* d_in, const int* in_sizes, int n_in,
                              void* d_out, int out_size, void* d_ws, size_t ws_size,
                              hipStream_t stream) {
  (void)n_in; (void)out_size; (void)ws_size;
  const float* x_user   = (const float*)d_in[0];
  const float* x_movie  = (const float*)d_in[1];
  const float* x_dir    = (const float*)d_in[2];
  const int*   rates_src = (const int*)d_in[3];
  const int*   rates_dst = (const int*)d_in[4];
  const int*   rby_src   = (const int*)d_in[5];
  const int*   rby_dst   = (const int*)d_in[6];
  const int*   dir_src   = (const int*)d_in[7];
  const int*   dir_dst   = (const int*)d_in[8];
  const int*   dby_src   = (const int*)d_in[9];
  const int*   dby_dst   = (const int*)d_in[10];
  const float* W_user  = (const float*)d_in[11];
  const float* b_user  = (const float*)d_in[12];
  const float* W_movie = (const float*)d_in[13];
  const float* b_movie = (const float*)d_in[14];
  const float* W_dir   = (const float*)d_in[15];
  const float* b_dir   = (const float*)d_in[16];
  const float* W_out   = (const float*)d_in[17];
  const float* b_out   = (const float*)d_in[18];

  const int F       = 128;
  const int n_user  = in_sizes[0] / F;
  const int n_movie = in_sizes[1] / F;
  const int n_dir   = in_sizes[2] / F;
  const int E_rates = in_sizes[3];
  const int E_rby   = in_sizes[5];
  const int E_dir   = in_sizes[7];
  const int E_dby   = in_sizes[9];
  const int HID     = in_sizes[11] / F;     // 128
  const int OUTD    = in_sizes[17] / HID;   // 16

  // ---- workspace carve-out (256B aligned) ----
  char* ws = (char*)d_ws;
  size_t off = 0;
  auto alloc_f = [&](size_t elems) -> float* {
    float* p = (float*)(ws + off);
    off += ((elems * sizeof(float)) + 255) & ~(size_t)255;
    return p;
  };
  float* h_u = alloc_f((size_t)n_user  * HID);
  float* h_m = alloc_f((size_t)n_movie * HID);
  float* h_d = alloc_f((size_t)n_dir   * HID);
  size_t zero_begin = off;
  float* agg_u = alloc_f((size_t)n_user  * HID);
  float* agg_m = alloc_f((size_t)n_movie * HID);
  float* agg_d = alloc_f((size_t)n_dir   * HID);
  int* deg_u  = (int*)alloc_f((size_t)n_user);
  int* deg_mr = (int*)alloc_f((size_t)n_movie);
  int* deg_md = (int*)alloc_f((size_t)n_movie);
  int* deg_dd = (int*)alloc_f((size_t)n_dir);
  size_t zero_end = off;

  // 1) zero agg accumulators + degree counters (contiguous region, float4 stores)
  long n4z = (long)(zero_end - zero_begin) / 16;
  zero4_kernel<<<cdiv_i(n4z, 256), 256, 0, stream>>>((float4*)(ws + zero_begin), n4z);

  // 2) per-etype in-degrees (computed once; reused by both layers)
  degree_kernel<<<cdiv_i(E_rates, 256), 256, 0, stream>>>(rates_dst, deg_mr, E_rates);
  degree_kernel<<<cdiv_i(E_dir,   256), 256, 0, stream>>>(dir_dst,   deg_md, E_dir);
  degree_kernel<<<cdiv_i(E_rby,   256), 256, 0, stream>>>(rby_dst,   deg_u,  E_rby);
  degree_kernel<<<cdiv_i(E_dby,   256), 256, 0, stream>>>(dby_dst,   deg_dd, E_dby);

  // 3) input projections + ReLU via bf16 WMMA
  auto launch_gemm = [&](const float* Ain, const float* Wm, const float* bb, float* Cout,
                         int M, int N, int K, bool relu) {
    int waves  = (M >> 4) * (N >> 4);
    int blocks = cdiv_i(waves, 8);       // 8 waves (256 threads) per block
    if (relu)
      gemm_bias_wmma_kernel<true ><<<blocks, 256, 0, stream>>>(Ain, Wm, bb, Cout, M, N, K);
    else
      gemm_bias_wmma_kernel<false><<<blocks, 256, 0, stream>>>(Ain, Wm, bb, Cout, M, N, K);
  };
  launch_gemm(x_user,  W_user,  b_user,  h_u, n_user,  HID, F, true);
  launch_gemm(x_movie, W_movie, b_movie, h_m, n_movie, HID, F, true);
  launch_gemm(x_dir,   W_dir,   b_dir,   h_d, n_dir,   HID, F, true);

  // 4) two propagation layers: scatter normalized means, then combine (which re-zeros agg)
  for (int layer = 0; layer < 2; ++layer) {
    edge_scatter_kernel<<<cdiv_i(E_rates, 8), 256, 0, stream>>>(h_u, rates_src, rates_dst, deg_mr, agg_m, E_rates, 0.5f);
    edge_scatter_kernel<<<cdiv_i(E_dir,   8), 256, 0, stream>>>(h_d, dir_src,   dir_dst,   deg_md, agg_m, E_dir,   0.5f);
    edge_scatter_kernel<<<cdiv_i(E_rby,   8), 256, 0, stream>>>(h_m, rby_src,   rby_dst,   deg_u,  agg_u, E_rby,   1.0f);
    edge_scatter_kernel<<<cdiv_i(E_dby,   8), 256, 0, stream>>>(h_m, dby_src,   dby_dst,   deg_dd, agg_d, E_dby,   1.0f);

    long nu4 = (long)n_user  * HID / 4;
    long nm4 = (long)n_movie * HID / 4;
    long nd4 = (long)n_dir   * HID / 4;
    combine_kernel<<<cdiv_i(nu4, 256), 256, 0, stream>>>((float4*)h_u, (float4*)agg_u, nu4, ALPHA_HSGC);
    combine_kernel<<<cdiv_i(nm4, 256), 256, 0, stream>>>((float4*)h_m, (float4*)agg_m, nm4, ALPHA_HSGC);
    combine_kernel<<<cdiv_i(nd4, 256), 256, 0, stream>>>((float4*)h_d, (float4*)agg_d, nd4, ALPHA_HSGC);
  }

  // 5) output head: logits[n_movie, 16] = h_m @ W_out^T + b_out
  launch_gemm(h_m, W_out, b_out, (float*)d_out, n_movie, OUTD, HID, false);
}
